// SimpleGRU_83081847374005
// MI455X (gfx1250) — compile-verified, matrix operand order checked by reading, not run on hardware
//
#include <hip/hip_runtime.h>

#define BB 64
#define TT 512
#define II 512
#define HH 1024
#define OO 512

typedef __attribute__((ext_vector_type(16))) __bf16 v16bf;
typedef __attribute__((ext_vector_type(8)))  float  v8f;
typedef unsigned short u16;

// ---------- helpers ----------

__device__ __forceinline__ u16 f2bf(float f) {
  unsigned int u = __float_as_uint(f);
  u += 0x7FFFu + ((u >> 16) & 1u);          // round-to-nearest-even
  return (u16)(u >> 16);
}

// fp32 -> bf16 grid-stride converter (one-time prepass; weights + source)
__global__ __launch_bounds__(256) void cvt_f32_to_bf16(const float* __restrict__ src,
                                                       u16* __restrict__ dst, long n) {
  long i = (long)blockIdx.x * blockDim.x + threadIdx.x;
  long stride = (long)gridDim.x * blockDim.x;
  for (; i < n; i += stride) dst[i] = f2bf(src[i]);
}

// Load one 16x32 bf16 WMMA fragment from a row-major [rows x ld] bf16 matrix.
// A-matrix layout (ISA 7.12.2): lane<16 holds row (row0+lane), K = k0..k0+7 and
// k0+16..k0+23; lane>=16 holds the k0+8.. / k0+24.. runs. Identical pattern works
// for the B matrix when the weight is stored [N x K] row-major (PyTorch layout),
// since B wants lane n to hold column n's K-run == W row n's K-run.
__device__ __forceinline__ v16bf load_frag(const u16* __restrict__ base, long ld,
                                           int row0, int k0) {
  int lane = (int)(threadIdx.x & 31u);
  const u16* p = base + (long)(row0 + (lane & 15)) * ld + k0 + ((lane >> 4) << 3);
  union { uint4 q[2]; v16bf v; } u;
  u.q[0] = *(const uint4*)p;        // K = k .. k+7   (x2 bf16 per dword)
  u.q[1] = *(const uint4*)(p + 16); // K = k+16 .. k+23
  return u.v;
}

__device__ __forceinline__ v8f wmma_bf16(v16bf a, v16bf b, v8f c) {
  // D = A(16x32) x B(32x16) + C, fp32 accumulate
  return __builtin_amdgcn_wmma_f32_16x16x32_bf16(false, a, false, b, (short)0, c,
                                                 false, false);
}

__device__ __forceinline__ float sigmoidf_(float x) {
  return 1.0f / (1.0f + __expf(-x));
}

// ---------- fused GRU cell: one block owns 16 columns of H, all 6 gate tiles ----------
// block = 128 threads = 4 waves; wave w computes M-rows [16w, 16w+16)
__global__ __launch_bounds__(128) void gru_cell_kernel(
    const u16* __restrict__ xB, long x_ld, int Kin,   // input activations, bf16 [64 x Kin]
    const u16* __restrict__ hB,                       // prev h, bf16 [64 x HH]
    const float* __restrict__ hF,                     // prev h, fp32 [64 x HH]
    const u16* __restrict__ Wih,                      // [3H x Kin] bf16, row-major
    const u16* __restrict__ Whh,                      // [3H x HH]  bf16, row-major
    const float* __restrict__ bih, const float* __restrict__ bhh,
    float* __restrict__ hFo, u16* __restrict__ hBo) { // new h (fp32 + bf16)
  const int lane = (int)(threadIdx.x & 31u);
  const int m0   = (int)(threadIdx.x >> 5) * 16;      // wave id -> M tile
  const int col0 = blockIdx.x * 16;                   // 16 H-columns per block

  v8f zero = {0.f, 0.f, 0.f, 0.f, 0.f, 0.f, 0.f, 0.f};
  v8f accI[3] = {zero, zero, zero};                   // gi_r, gi_z, gi_n
  v8f accH[3] = {zero, zero, zero};                   // gh_r, gh_z, gh_n

  // gi = x @ W_ih^T  (three gate row-blocks of W_ih)
  for (int k0 = 0; k0 < Kin; k0 += 32) {
    v16bf a = load_frag(xB, x_ld, m0, k0);
#pragma unroll
    for (int g = 0; g < 3; ++g) {
      v16bf b = load_frag(Wih, Kin, g * HH + col0, k0);
      accI[g] = wmma_bf16(a, b, accI[g]);
    }
  }
  // gh = h @ W_hh^T
  for (int k0 = 0; k0 < HH; k0 += 32) {
    v16bf a = load_frag(hB, HH, m0, k0);
#pragma unroll
    for (int g = 0; g < 3; ++g) {
      v16bf b = load_frag(Whh, HH, g * HH + col0, k0);
      accH[g] = wmma_bf16(a, b, accH[g]);
    }
  }

  // Epilogue: C layout -> element i of v8f at lane l is (m = m0+i+8*(l>>4), n = col0+(l&15))
  const int col = col0 + (lane & 15);
  const float br = bih[col],          cr = bhh[col];
  const float bz = bih[HH + col],     cz = bhh[HH + col];
  const float bn = bih[2 * HH + col], cn = bhh[2 * HH + col];
#pragma unroll
  for (int i = 0; i < 8; ++i) {
    const int m = m0 + i + ((lane >> 4) << 3);
    const float r  = sigmoidf_((accI[0][i] + br) + (accH[0][i] + cr));
    const float z  = sigmoidf_((accI[1][i] + bz) + (accH[1][i] + cz));
    const float nn = tanhf((accI[2][i] + bn) + r * (accH[2][i] + cn));
    const float hold = hF[(long)m * HH + col];
    const float hnew = (1.0f - z) * nn + z * hold;
    hFo[(long)m * HH + col] = hnew;
    hBo[(long)m * HH + col] = f2bf(hnew);
  }
}

// ---------- output projection: out[t] = h1 @ W_out^T + b_out ----------
__global__ __launch_bounds__(128) void out_proj_kernel(
    const u16* __restrict__ hB,         // [64 x HH] bf16
    const u16* __restrict__ Wout,       // [OO x HH] bf16, row-major
    const float* __restrict__ bout,
    float* __restrict__ out) {          // [64 x OO] fp32 (one timestep)
  const int lane = (int)(threadIdx.x & 31u);
  const int m0   = (int)(threadIdx.x >> 5) * 16;
  const int col0 = blockIdx.x * 16;

  v8f acc = {0.f, 0.f, 0.f, 0.f, 0.f, 0.f, 0.f, 0.f};
  for (int k0 = 0; k0 < HH; k0 += 32) {
    v16bf a = load_frag(hB, HH, m0, k0);
    v16bf b = load_frag(Wout, HH, col0, k0);
    acc = wmma_bf16(a, b, acc);
  }

  const int col = col0 + (lane & 15);
  const float bo = bout[col];
#pragma unroll
  for (int i = 0; i < 8; ++i) {
    const int m = m0 + i + ((lane >> 4) << 3);
    out[(long)m * OO + col] = acc[i] + bo;
  }
}

// ---------- host launcher ----------

extern "C" void kernel_launch(void* const* d_in, const int* in_sizes, int n_in,
                              void* d_out, int out_size, void* d_ws, size_t ws_size,
                              hipStream_t stream) {
  (void)in_sizes; (void)n_in; (void)out_size; (void)ws_size;
  const float* src  = (const float*)d_in[0];
  // d_in[1] = target (unused by reference)
  const float* Wih0 = (const float*)d_in[2];
  const float* Whh0 = (const float*)d_in[3];
  const float* bih0 = (const float*)d_in[4];
  const float* bhh0 = (const float*)d_in[5];
  const float* Wih1 = (const float*)d_in[6];
  const float* Whh1 = (const float*)d_in[7];
  const float* bih1 = (const float*)d_in[8];
  const float* bhh1 = (const float*)d_in[9];
  const float* Wout = (const float*)d_in[10];
  const float* bout = (const float*)d_in[11];
  float* out = (float*)d_out;

  // workspace layout (bf16 region first, then fp32 h carries)
  u16* ws = (u16*)d_ws;
  const size_t n_src  = (size_t)BB * TT * II;     // 16,777,216
  const size_t n_wih0 = (size_t)3 * HH * II;      //  1,572,864
  const size_t n_whh  = (size_t)3 * HH * HH;      //  3,145,728
  const size_t n_wout = (size_t)OO * HH;          //    524,288
  const size_t n_h    = (size_t)BB * HH;          //     65,536

  u16* srcb  = ws;
  u16* Wih0b = srcb  + n_src;
  u16* Whh0b = Wih0b + n_wih0;
  u16* Wih1b = Whh0b + n_whh;
  u16* Whh1b = Wih1b + n_whh;
  u16* Woutb = Whh1b + n_whh;
  u16* h0b[2] = {Woutb + n_wout, Woutb + n_wout + n_h};
  u16* h1b[2] = {h0b[1] + n_h, h0b[1] + 2 * n_h};
  float* fbase = (float*)(h1b[1] + n_h);
  float* h0f[2] = {fbase, fbase + n_h};
  float* h1f[2] = {fbase + 2 * n_h, fbase + 3 * n_h};

  // one-time prepass: weights + source -> bf16; zero t=0 hidden state
  cvt_f32_to_bf16<<<2048, 256, 0, stream>>>(src,  srcb,  (long)n_src);
  cvt_f32_to_bf16<<<1024, 256, 0, stream>>>(Wih0, Wih0b, (long)n_wih0);
  cvt_f32_to_bf16<<<1024, 256, 0, stream>>>(Whh0, Whh0b, (long)n_whh);
  cvt_f32_to_bf16<<<1024, 256, 0, stream>>>(Wih1, Wih1b, (long)n_whh);
  cvt_f32_to_bf16<<<1024, 256, 0, stream>>>(Whh1, Whh1b, (long)n_whh);
  cvt_f32_to_bf16<<<1024, 256, 0, stream>>>(Wout, Woutb, (long)n_wout);
  hipMemsetAsync(h0b[0], 0, n_h * sizeof(u16), stream);
  hipMemsetAsync(h1b[0], 0, n_h * sizeof(u16), stream);
  hipMemsetAsync(h0f[0], 0, n_h * sizeof(float), stream);
  hipMemsetAsync(h1f[0], 0, n_h * sizeof(float), stream);

  const dim3 blk(128);
  const dim3 gridCell(HH / 16);   // 64 blocks
  const dim3 gridProj(OO / 16);   // 32 blocks

  for (int t = 0; t < TT; ++t) {
    const int pr = t & 1, cu = pr ^ 1;
    const u16* xB = srcb + (size_t)t * II;      // row stride = TT*II (source is [B,T,I])
    gru_cell_kernel<<<gridCell, blk, 0, stream>>>(
        xB, (long)TT * II, II,
        h0b[pr], h0f[pr], Wih0b, Whh0b, bih0, bhh0,
        h0f[cu], h0b[cu]);
    gru_cell_kernel<<<gridCell, blk, 0, stream>>>(
        h0b[cu], (long)HH, HH,
        h1b[pr], h1f[pr], Wih1b, Whh1b, bih1, bhh1,
        h1f[cu], h1b[cu]);
    out_proj_kernel<<<gridProj, blk, 0, stream>>>(
        h1b[cu], Woutb, bout, out + (size_t)t * BB * OO);
  }
}
